// causalAttention_35777077576384
// MI455X (gfx1250) — matrix operand order repaired and checked by reading, not compiled, not run on hardware
//
#include <hip/hip_runtime.h>
#include <hip/hip_bf16.h>
#include <math.h>

// ---------------- types ----------------
typedef _Float16 half_t;
typedef half_t v16h __attribute__((ext_vector_type(16)));
typedef half_t v8h  __attribute__((ext_vector_type(8)));
typedef float  v8f  __attribute__((ext_vector_type(8)));

#define EMBED   1024
#define NHEADS  16
#define HDIM    64
#define BATCH   2
#define SEQ     2048
#define MROWS   (BATCH * SEQ)   // 4096
#define N3C     (3 * EMBED)     // 3072

// ---------------- WMMA helpers ----------------
__device__ __forceinline__ v8f wmma16(v16h a, v16h b, v8f c) {
  // D = A(16x32 f16) * B(32x16 f16) + C(16x16 f32)
  return __builtin_amdgcn_wmma_f32_16x16x32_f16(false, a, false, b, (short)0, c,
                                                false, false);
}

// A-fragment (16x32, row-major source). Per-lane layout:
// lane = m + 16*g ; element e holds K = 16*(e/8) + 8*g + (e%8)
// -> two contiguous 16-byte loads per lane.
__device__ __forceinline__ v16h load_a(const half_t* base, int ld, int row0, int col0) {
  const int l = threadIdx.x & 31, m = l & 15, g = l >> 4;
  const half_t* p = base + (size_t)(row0 + m) * ld + col0 + 8 * g;
  v8h lo = *(const v8h*)p;          // K = 8g .. 8g+7
  v8h hi = *(const v8h*)(p + 16);   // K = 16+8g .. 16+8g+7
  return __builtin_shufflevector(lo, hi, 0, 1, 2, 3, 4, 5, 6, 7,
                                 8, 9, 10, 11, 12, 13, 14, 15);
}

// B-fragment (32x16) from B^T stored row-major [N][K]. Per-lane layout:
// lane = n + 16*g ; element e holds K = 16*(e/8) + 8*g + (e%8) -> two 16-byte loads.
__device__ __forceinline__ v16h load_b(const half_t* bt, int ld, int n0, int k0) {
  const int l = threadIdx.x & 31, n = l & 15, g = l >> 4;
  const half_t* p = bt + (size_t)(n0 + n) * ld + k0 + 8 * g;
  v8h lo = *(const v8h*)p;
  v8h hi = *(const v8h*)(p + 16);
  return __builtin_shufflevector(lo, hi, 0, 1, 2, 3, 4, 5, 6, 7,
                                 8, 9, 10, 11, 12, 13, 14, 15);
}

// 16-lane-group reductions (wave32: xor masks 1,2,4,8 stay inside each half)
__device__ __forceinline__ float rmax16(float v) {
  v = fmaxf(v, __shfl_xor(v, 1));
  v = fmaxf(v, __shfl_xor(v, 2));
  v = fmaxf(v, __shfl_xor(v, 4));
  v = fmaxf(v, __shfl_xor(v, 8));
  return v;
}
__device__ __forceinline__ float rsum16(float v) {
  v += __shfl_xor(v, 1);
  v += __shfl_xor(v, 2);
  v += __shfl_xor(v, 4);
  v += __shfl_xor(v, 8);
  return v;
}

// ---------------- prep kernels ----------------
__global__ void __launch_bounds__(256) k_cvt_f32_f16(const float* __restrict__ src,
                                                     half_t* __restrict__ dst, size_t n) {
  size_t i = (size_t)blockIdx.x * blockDim.x + threadIdx.x;
  if (i < n) dst[i] = (half_t)src[i];
}

// w [K][N] f32 -> wt [N][K] f16
__global__ void __launch_bounds__(256) k_transpose_w(const float* __restrict__ w,
                                                     half_t* __restrict__ wt,
                                                     int K, int N) {
  size_t i = (size_t)blockIdx.x * blockDim.x + threadIdx.x;
  if (i >= (size_t)K * N) return;
  int n = (int)(i / K), k = (int)(i % K);
  wt[i] = (half_t)w[(size_t)k * N + n];
}

// vt[b][h][d][t] = kqv[b*SEQ+t][2*EMBED + h*HDIM + d]   (v is third in k,q,v split)
__global__ void __launch_bounds__(256) k_build_vt(const half_t* __restrict__ kqv,
                                                  half_t* __restrict__ vt) {
  size_t i = (size_t)blockIdx.x * blockDim.x + threadIdx.x;
  if (i >= (size_t)BATCH * NHEADS * HDIM * SEQ) return;
  int t = (int)(i & (SEQ - 1));
  size_t r = i >> 11;
  int d = (int)(r & (HDIM - 1)); r >>= 6;
  int h = (int)(r & (NHEADS - 1));
  int b = (int)(r >> 4);
  vt[i] = kqv[((size_t)b * SEQ + t) * N3C + 2 * EMBED + h * HDIM + d];
}

// ---------------- generic WMMA GEMM: C[M][N] = A[M][K] * B, B given as B^T[N][K] ----------------
template <bool OUT_F32>
__global__ void __launch_bounds__(256) gemm_wmma(const half_t* __restrict__ A,
                                                 const half_t* __restrict__ BT,
                                                 void* __restrict__ Cout,
                                                 int Md, int Nd, int Kd) {
  const int l = threadIdx.x & 31, m = l & 15, g = l >> 4;
  const int w = blockIdx.x * (blockDim.x >> 5) + (threadIdx.x >> 5);
  const int tilesN = Nd >> 6;
  const int tm = w / tilesN, tn = w % tilesN;     // 32(M) x 64(N) tile per wave
  if (tm * 32 >= Md) return;                      // wave-uniform guard

  v8f acc[2][4];
#pragma unroll
  for (int i = 0; i < 2; ++i)
#pragma unroll
    for (int f = 0; f < 4; ++f)
      acc[i][f] = (v8f){0.f, 0.f, 0.f, 0.f, 0.f, 0.f, 0.f, 0.f};

  for (int k0 = 0; k0 < Kd; k0 += 32) {
    v16h a0 = load_a(A, Kd, tm * 32, k0);
    v16h a1 = load_a(A, Kd, tm * 32 + 16, k0);
#pragma unroll
    for (int f = 0; f < 4; ++f) {
      v16h bfr = load_b(BT, Kd, tn * 64 + 16 * f, k0);
      acc[0][f] = wmma16(a0, bfr, acc[0][f]);
      acc[1][f] = wmma16(a1, bfr, acc[1][f]);
    }
  }

  // C/D layout: reg r, lane l : row = r + 8*(l/16), col = l%16
#pragma unroll
  for (int i = 0; i < 2; ++i)
#pragma unroll
    for (int f = 0; f < 4; ++f)
#pragma unroll
      for (int r = 0; r < 8; ++r) {
        size_t row = (size_t)tm * 32 + i * 16 + r + 8 * g;
        size_t col = (size_t)tn * 64 + f * 16 + m;
        if (OUT_F32)
          ((float*)Cout)[row * Nd + col] = acc[i][f][r];
        else
          ((half_t*)Cout)[row * Nd + col] = (half_t)acc[i][f][r];
      }
}

// ---------------- flash attention: one wave per (b, h, 16-row q block) ----------------
__global__ void __launch_bounds__(256) attn_kernel(const half_t* __restrict__ kqv,
                                                   const half_t* __restrict__ vt,
                                                   half_t* __restrict__ y) {
  __shared__ half_t tile[8][16][32];            // per-wave P staging (f16)
  const int wslab = threadIdx.x >> 5;
  const int l = threadIdx.x & 31, m = l & 15, g = l >> 4;
  const int w = blockIdx.x * 8 + wslab;
  const int qblk = w & 127;                     // 128 q-blocks of 16 rows
  const int bh = w >> 7;
  const int b = bh >> 4, h = bh & 15;

  // reference unpacks k,q,v in that order: k @ 0, q @ EMBED, v @ 2*EMBED
  const half_t* qbase = kqv + (size_t)b * SEQ * N3C + EMBED + h * HDIM;
  const half_t* kbase = kqv + (size_t)b * SEQ * N3C + h * HDIM;
  const half_t* vtb   = vt + (size_t)bh * HDIM * SEQ;   // [64][SEQ]

  // Q tile 16x64 -> two A fragments (K chunks of 32)
  v16h q0 = load_a(qbase, N3C, qblk * 16, 0);
  v16h q1 = load_a(qbase, N3C, qblk * 16, 32);

  v8f o[4];
  float m_i[8], l_i[8], alpha[8];
#pragma unroll
  for (int f = 0; f < 4; ++f) o[f] = (v8f){0.f, 0.f, 0.f, 0.f, 0.f, 0.f, 0.f, 0.f};
#pragma unroll
  for (int r = 0; r < 8; ++r) { m_i[r] = -INFINITY; l_i[r] = 0.f; }

  const int jend = qblk * 16 + 16;              // keys needed (causal)
  for (int j = 0; j < jend; j += 32) {
    // S = Q * K^T for 32 keys -> two 16x16 f32 fragments
    v8f s0 = (v8f){0.f, 0.f, 0.f, 0.f, 0.f, 0.f, 0.f, 0.f};
    v8f s1 = s0;
    v16h kf;
    kf = load_b(kbase, N3C, j, 0);       s0 = wmma16(q0, kf, s0);
    kf = load_b(kbase, N3C, j, 32);      s0 = wmma16(q1, kf, s0);
    kf = load_b(kbase, N3C, j + 16, 0);  s1 = wmma16(q0, kf, s1);
    kf = load_b(kbase, N3C, j + 16, 32); s1 = wmma16(q1, kf, s1);

    // scale, causal mask, online softmax; write P (f16) to wave-private LDS
#pragma unroll
    for (int r = 0; r < 8; ++r) {
      const int row = r + 8 * g;                // 0..15 within q block
      const int qi = qblk * 16 + row;
      float v0 = s0[r] * 0.125f;                // 1/sqrt(64)
      float v1 = s1[r] * 0.125f;
      if (j + m > qi)       v0 = -INFINITY;
      if (j + 16 + m > qi)  v1 = -INFINITY;
      float mx = rmax16(fmaxf(v0, v1));
      float mnew = fmaxf(m_i[r], mx);
      float al = __expf(m_i[r] - mnew);
      float p0 = __expf(v0 - mnew);
      float p1 = __expf(v1 - mnew);
      l_i[r] = l_i[r] * al + rsum16(p0 + p1);
      m_i[r] = mnew;
      alpha[r] = al;
      tile[wslab][row][m]      = (half_t)p0;
      tile[wslab][row][m + 16] = (half_t)p1;
    }

    // rescale O by alpha (per row)
#pragma unroll
    for (int f = 0; f < 4; ++f)
#pragma unroll
      for (int r = 0; r < 8; ++r) o[f][r] *= alpha[r];

    // ensure LDS stores complete before transpose-pattern reads (same wave)
    asm volatile("s_wait_dscnt 0" ::: "memory");

    // P (16x32) as A fragment from LDS
    const half_t* lp = &tile[wslab][m][8 * g];
    v8h plo = *(const v8h*)lp;
    v8h phi = *(const v8h*)(lp + 16);
    v16h pf = __builtin_shufflevector(plo, phi, 0, 1, 2, 3, 4, 5, 6, 7,
                                      8, 9, 10, 11, 12, 13, 14, 15);

    // O += P * V : V block (32 keys x 64 d) via V^T rows (contiguous per lane)
#pragma unroll
    for (int f = 0; f < 4; ++f) {
      v16h vb = load_b(vtb, SEQ, f * 16, j);
      o[f] = wmma16(pf, vb, o[f]);
    }
  }

  // normalize and store y[b, t, h*64 + d] as f16 for the output projection
#pragma unroll
  for (int f = 0; f < 4; ++f)
#pragma unroll
    for (int r = 0; r < 8; ++r) {
      size_t row = (size_t)b * SEQ + qblk * 16 + r + 8 * g;
      size_t col = (size_t)h * HDIM + f * 16 + m;
      y[row * EMBED + col] = (half_t)(o[f][r] / l_i[r]);
    }
}

// ---------------- host launch ----------------
extern "C" void kernel_launch(void* const* d_in, const int* in_sizes, int n_in,
                              void* d_out, int out_size, void* d_ws, size_t ws_size,
                              hipStream_t stream) {
  (void)in_sizes; (void)n_in; (void)out_size; (void)ws_size;
  const float* x      = (const float*)d_in[0];   // [B,T,C]
  const float* w_attn = (const float*)d_in[1];   // [C,3C]
  const float* w_proj = (const float*)d_in[2];   // [C,C]
  float* out = (float*)d_out;                    // [B,T,C]

  // workspace carve-up (f16 elements)
  half_t* xh   = (half_t*)d_ws;                          // 4096*1024
  half_t* watT = xh   + (size_t)MROWS * EMBED;           // 3072*1024 (w_attn^T)
  half_t* wpT  = watT + (size_t)N3C * EMBED;             // 1024*1024 (w_proj^T)
  half_t* kqvh = wpT  + (size_t)EMBED * EMBED;           // 4096*3072
  half_t* vth  = kqvh + (size_t)MROWS * N3C;             // 32*64*2048 (V^T per head)
  half_t* yh   = vth  + (size_t)BATCH * NHEADS * HDIM * SEQ; // 4096*1024

  const size_t nX = (size_t)MROWS * EMBED;
  k_cvt_f32_f16<<<(int)((nX + 255) / 256), 256, 0, stream>>>(x, xh, nX);

  k_transpose_w<<<(N3C * EMBED + 255) / 256, 256, 0, stream>>>(w_attn, watT, EMBED, N3C);
  k_transpose_w<<<(EMBED * EMBED + 255) / 256, 256, 0, stream>>>(w_proj, wpT, EMBED, EMBED);

  // kqv = x @ w_attn : 4096x3072x1024, 128*48 = 6144 wave-tiles, 8 waves/block
  gemm_wmma<false><<<768, 256, 0, stream>>>(xh, watT, (void*)kqvh, MROWS, N3C, EMBED);

  const size_t nVT = (size_t)BATCH * NHEADS * HDIM * SEQ;
  k_build_vt<<<(int)((nVT + 255) / 256), 256, 0, stream>>>(kqvh, vth);

  // flash attention: 32 (b,h) * 128 q-blocks = 4096 waves
  attn_kernel<<<512, 256, 0, stream>>>(kqvh, vth, yh);

  // out = y @ w_proj : 4096x1024x1024, 128*16 = 2048 wave-tiles
  gemm_wmma<true><<<256, 256, 0, stream>>>(yh, wpT, (void*)out, MROWS, EMBED, EMBED);
}
